// PropertyPredictorGNN_46316927320456
// MI455X (gfx1250) — matrix operand order repaired and checked by reading, not compiled, not run on hardware
//
#include <hip/hip_runtime.h>
#include <hip/hip_bf16.h>

#define N_NODES 8192
#define HIDDEN  256
#define OUT_DIM 16
#define NWORDS  (N_NODES / 32)   // 256 bitmask words per row

// Async global->LDS staging of the B panel (CDNA5 GLOBAL_LOAD_ASYNC_TO_LDS_B128,
// tracked by ASYNCcnt). Set to 0 to fall back to plain loads + ds stores.
#define USE_ASYNC_B 1

typedef __attribute__((ext_vector_type(16))) _Float16 v16h;
typedef __attribute__((ext_vector_type(8)))  _Float16 v8h;
typedef __attribute__((ext_vector_type(8)))  float    v8f;

__device__ __forceinline__ v16h cat16(v8h lo, v8h hi) {
    return __builtin_shufflevector(lo, hi, 0,1,2,3,4,5,6,7,8,9,10,11,12,13,14,15);
}

// ---------------------------------------------------------------------------
// K1: degree + binarized adjacency bitmask.  dinv[i] = rsqrt(1 + #neighbors).
// The single full 256MB read of the fp32 adjacency (~11us at 23.3 TB/s);
// emits an 8MB bitmask so no later pass touches the fp32 matrix again.
// ---------------------------------------------------------------------------
__global__ void __launch_bounds__(256) deg_bits_kernel(
        const float* __restrict__ adj, float* __restrict__ dinv,
        unsigned* __restrict__ bits) {
    const int wave = threadIdx.x >> 5, lane = threadIdx.x & 31;
    const int row = blockIdx.x * 8 + wave;
    const float4* rp = (const float4*)(adj + (size_t)row * (size_t)N_NODES);
    float cnt = 0.f;
    for (int w = lane; w < NWORDS; w += 32) {
        unsigned m = 0;
        #pragma unroll
        for (int q = 0; q < 8; ++q) {          // 8 float4 = 32 columns
            float4 v = rp[w * 8 + q];
            if (v.x != 0.f) m |= 1u << (q * 4 + 0);
            if (v.y != 0.f) m |= 1u << (q * 4 + 1);
            if (v.z != 0.f) m |= 1u << (q * 4 + 2);
            if (v.w != 0.f) m |= 1u << (q * 4 + 3);
        }
        bits[(size_t)row * NWORDS + w] = m;
        cnt += (float)__popc(m);
    }
    for (int o = 16; o; o >>= 1) cnt += __shfl_down(cnt, o, 32);
    if (lane == 0) dinv[row] = rsqrtf(cnt + 1.0f);   // +1 = self loop
}

// ---------------------------------------------------------------------------
// K2: s[i] = dinv_i * ( sum_{j in N(i)} dinv_j + dinv_i )  (row sums of Nmat)
// ---------------------------------------------------------------------------
__global__ void __launch_bounds__(256) s_kernel(
        const unsigned* __restrict__ bits, const float* __restrict__ dinv,
        float* __restrict__ s) {
    const int wave = threadIdx.x >> 5, lane = threadIdx.x & 31;
    const int row = blockIdx.x * 8 + wave;
    float t = 0.f;
    for (int w = lane; w < NWORDS; w += 32) {
        unsigned m = bits[(size_t)row * NWORDS + w];
        while (m) {
            int b = __ffs(m) - 1;
            m &= m - 1;
            t += dinv[w * 32 + b];
        }
    }
    for (int o = 16; o; o >>= 1) t += __shfl_down(t, o, 32);
    if (lane == 0) { float di = dinv[row]; s[row] = di * (t + di); }
}

// K3: h1[i,f] = relu(s_i * W1[f] + b1[f]) -> f16 (x = ones makes L1 rank-1)
__global__ void h1_kernel(const float* __restrict__ s, const float* __restrict__ W1,
                          const float* __restrict__ b1, _Float16* __restrict__ h1) {
    const int i = blockIdx.x, f = threadIdx.x;
    h1[(size_t)i * HIDDEN + f] = (_Float16)fmaxf(0.f, s[i] * W1[f] + b1[f]);
}

// K3b: W2 (row-major f32) -> W2t (transposed f16) for contiguous B fragments
__global__ void w2t_kernel(const float* __restrict__ W2, _Float16* __restrict__ W2t) {
    const int k = blockIdx.x, n = threadIdx.x;
    W2t[(size_t)n * HIDDEN + k] = (_Float16)W2[(size_t)k * HIDDEN + n];
}

__global__ void zero_kernel(float* __restrict__ p) { p[threadIdx.x] = 0.f; }

// ---------------------------------------------------------------------------
// K4: M = h1 @ W2 (WMMA f16, f32 acc); epilogue writes Pt[f][j] = dinv_j*M[j,f]
// (f16, transposed) so the big GEMM's B panel is contiguous per column.
// ---------------------------------------------------------------------------
__global__ void __launch_bounds__(256) gemm_h1w2_kernel(
        const _Float16* __restrict__ h1, const _Float16* __restrict__ W2t,
        const float* __restrict__ dinv, _Float16* __restrict__ Pt) {
    const int wave = threadIdx.x >> 5, lane = threadIdx.x & 31;
    const int rowtile = wave >> 2;            // 0..1
    const int colgrp  = wave & 3;             // 0..3 (64 cols each)
    const int m0 = blockIdx.x * 32 + rowtile * 16;
    const int mrow = m0 + (lane & 15);
    const int hi = lane >> 4;                 // K-half select per ISA A layout
    v8f acc[4] = {};
    for (int k0 = 0; k0 < HIDDEN; k0 += 32) {
        const _Float16* ap = h1 + (size_t)mrow * HIDDEN + k0 + hi * 8;
        v16h a = cat16(*(const v8h*)ap, *(const v8h*)(ap + 16));
        #pragma unroll
        for (int t = 0; t < 4; ++t) {
            const int n = colgrp * 64 + t * 16 + (lane & 15);
            const _Float16* bp = W2t + (size_t)n * HIDDEN + k0 + hi * 8;
            v16h b = cat16(*(const v8h*)bp, *(const v8h*)(bp + 16));
            acc[t] = __builtin_amdgcn_wmma_f32_16x16x32_f16(
                false, a, false, b, (short)0, acc[t], false, false);
        }
    }
    #pragma unroll
    for (int t = 0; t < 4; ++t) {
        const int n = colgrp * 64 + t * 16 + (lane & 15);
        #pragma unroll
        for (int v = 0; v < 8; ++v) {
            const int row = m0 + v + hi * 8;   // C/D layout: VGPR v -> M=v / v+8
            Pt[(size_t)n * N_NODES + row] = (_Float16)(acc[t][v] * dinv[row]);
        }
    }
}

// ---------------------------------------------------------------------------
// K5: big aggregation  Q = (A_bin + I) @ P  (M=8192, N=256, K=8192) WMMA f16.
// Block tile BM=64 x BN=128; 8 waves = 4 rowtiles x 2 colgroups.
//  - A tile (64x64): unpacked from the bitmask into LDS as exact {0,1} f16
//    with diagonal 1s (VALU transform).
//  - B panel (128 cols x 64 k): async-copied global->LDS once per k-tile and
//    shared by all 4 rowtile waves (4x less L2 traffic than per-wave loads).
// Fused epilogue: h2 = relu(dinv_i*Q + b2) is row-summed straight into
// `pooled` — h2 never touches memory.
// ---------------------------------------------------------------------------
__global__ void __launch_bounds__(256) gemm_agg_kernel(
        const unsigned* __restrict__ bits, const _Float16* __restrict__ Pt,
        const float* __restrict__ dinv, const float* __restrict__ b2,
        float* __restrict__ pooled) {
    __shared__ alignas(16) _Float16 lds_a[64][64];    //  8 KB: A tile {0,1}
    __shared__ alignas(16) _Float16 lds_b[128][64];   // 16 KB: B panel
    __shared__ float lds_pool[128];
    const int tid = threadIdx.x;
    const int wave = tid >> 5, lane = tid & 31;
    const int rowtile = wave >> 1;            // 0..3 (16 rows each)
    const int colgrp  = wave & 1;             // 0..1 (64 cols each)
    const int r0 = blockIdx.x * 64;
    const int n0 = blockIdx.y * 128;
    const int hi = lane >> 4;
    const int mloc = rowtile * 16 + (lane & 15);
    if (tid < 128) lds_pool[tid] = 0.f;

#if USE_ASYNC_B
    const unsigned ldsb_base = (unsigned)(uintptr_t)(&lds_b[0][0]);
#endif

    v8f acc[4] = {};
    for (int k0 = 0; k0 < N_NODES; k0 += 64) {
        __syncthreads();                       // drains DScnt: prior readers done
        // ---- stage A: each thread unpacks 16 bits -> 16 f16 {0,1} + diagonal
        {
            const int rr = tid >> 2;           // 0..63 tile row
            const int seg = tid & 3;           // 16-col segment
            const int grow = r0 + rr;
            const unsigned word =
                bits[(size_t)grow * NWORDS + (k0 >> 5) + (seg >> 1)];
            const unsigned half16 = (word >> (16 * (seg & 1))) & 0xFFFFu;
            const int cbase = seg * 16;
            #pragma unroll
            for (int i = 0; i < 16; ++i) {
                const int gc = k0 + cbase + i;
                lds_a[rr][cbase + i] =
                    (_Float16)(((half16 >> i) & 1u) || (gc == grow) ? 1.f : 0.f);
            }
            if (k0 + 64 < N_NODES)             // global_prefetch_b8 next bit tile
                __builtin_prefetch(&bits[(size_t)grow * NWORDS + ((k0 + 64) >> 5)], 0, 0);
        }
        // ---- stage B: copy Pt[n0..n0+127][k0..k0+63] (16 KB) into LDS
        #pragma unroll
        for (int j = 0; j < 4; ++j) {          // 1024 x 16B chunks / 256 threads
            const int q = tid + j * 256;
            const int brow = q >> 3;           // 0..127 panel row (column of P)
            const int c8 = (q & 7) * 8;        // 8-half chunk within the row
            const _Float16* gp = Pt + (size_t)(n0 + brow) * N_NODES + k0 + c8;
#if USE_ASYNC_B
            const unsigned ldst = ldsb_base + (unsigned)(brow * 64 + c8) * 2u;
            asm volatile("global_load_async_to_lds_b128 %0, %1, off"
                         :: "v"(ldst), "v"((unsigned long long)(uintptr_t)gp)
                         : "memory");
#else
            *(v8h*)&lds_b[brow][c8] = *(const v8h*)gp;
#endif
        }
#if USE_ASYNC_B
        asm volatile("s_wait_asynccnt 0x0" ::: "memory");
#endif
        __syncthreads();
        // ---- 2 WMMA k-steps of 32, all operands from LDS
        #pragma unroll
        for (int kk = 0; kk < 2; ++kk) {
            const int ks = kk * 32;
            const _Float16* ap = &lds_a[mloc][ks + hi * 8];      // ds_load_b128 x2
            v16h a = cat16(*(const v8h*)ap, *(const v8h*)(ap + 16));
            #pragma unroll
            for (int t = 0; t < 4; ++t) {
                const int lcol = colgrp * 64 + t * 16 + (lane & 15);
                const _Float16* bp = &lds_b[lcol][ks + hi * 8];
                v16h b = cat16(*(const v8h*)bp, *(const v8h*)(bp + 16));
                acc[t] = __builtin_amdgcn_wmma_f32_16x16x32_f16(
                    false, a, false, b, (short)0, acc[t], false, false);
            }
        }
    }
    // ---- fused epilogue: pooled[col] += sum_rows relu(dinv_row * Q + b2[col])
    float dv[8];
    #pragma unroll
    for (int v = 0; v < 8; ++v) dv[v] = dinv[r0 + rowtile * 16 + v + hi * 8];
    #pragma unroll
    for (int t = 0; t < 4; ++t) {
        const int lcol = colgrp * 64 + t * 16 + (lane & 15);
        const float bb = b2[n0 + lcol];
        float part = 0.f;
        #pragma unroll
        for (int v = 0; v < 8; ++v)
            part += fmaxf(0.f, dv[v] * acc[t][v] + bb);
        part += __shfl_xor(part, 16, 32);      // merge the two row halves
        if (lane < 16) atomicAdd(&lds_pool[lcol], part);
    }
    __syncthreads();
    if (tid < 128) atomicAdd(&pooled[n0 + tid], lds_pool[tid]);
}

// K6: out[o] = pooled @ Wfc + bfc  (tiny: 256x16)
__global__ void fc_kernel(const float* __restrict__ pooled, const float* __restrict__ Wfc,
                          const float* __restrict__ bfc, float* __restrict__ out) {
    const int o = threadIdx.x;
    if (o < OUT_DIM) {
        float acc = bfc[o];
        for (int f = 0; f < HIDDEN; ++f) acc += pooled[f] * Wfc[f * OUT_DIM + o];
        out[o] = acc;
    }
}

extern "C" void kernel_launch(void* const* d_in, const int* in_sizes, int n_in,
                              void* d_out, int out_size, void* d_ws, size_t ws_size,
                              hipStream_t stream) {
    (void)in_sizes; (void)n_in; (void)out_size; (void)ws_size;
    const float* adj = (const float*)d_in[0];
    const float* W1  = (const float*)d_in[1];
    const float* b1  = (const float*)d_in[2];
    const float* W2  = (const float*)d_in[3];
    const float* b2  = (const float*)d_in[4];
    const float* Wfc = (const float*)d_in[5];
    const float* bfc = (const float*)d_in[6];
    float* out = (float*)d_out;

    // Workspace carve-out (256B aligned): ~16.6 MB total
    char* ws = (char*)d_ws;
    size_t off = 0;
    auto carve = [&](size_t bytes) {
        char* p = ws + off;
        off = (off + bytes + 255) & ~(size_t)255;
        return p;
    };
    float*    dinv   = (float*)carve((size_t)N_NODES * 4);
    float*    s      = (float*)carve((size_t)N_NODES * 4);
    unsigned* bits   = (unsigned*)carve((size_t)N_NODES * NWORDS * 4);   // 8 MB
    _Float16* h1     = (_Float16*)carve((size_t)N_NODES * HIDDEN * 2);   // 4 MB
    _Float16* W2t    = (_Float16*)carve((size_t)HIDDEN * HIDDEN * 2);
    _Float16* Pt     = (_Float16*)carve((size_t)N_NODES * HIDDEN * 2);   // 4 MB
    float*    pooled = (float*)carve((size_t)HIDDEN * 4);

    deg_bits_kernel<<<N_NODES / 8, 256, 0, stream>>>(adj, dinv, bits);
    s_kernel<<<N_NODES / 8, 256, 0, stream>>>(bits, dinv, s);
    h1_kernel<<<N_NODES, HIDDEN, 0, stream>>>(s, W1, b1, h1);
    w2t_kernel<<<HIDDEN, HIDDEN, 0, stream>>>(W2, W2t);
    zero_kernel<<<1, HIDDEN, 0, stream>>>(pooled);
    gemm_h1w2_kernel<<<N_NODES / 32, 256, 0, stream>>>(h1, W2t, dinv, Pt);
    dim3 agrid(N_NODES / 64, 2);
    gemm_agg_kernel<<<agrid, 256, 0, stream>>>(bits, Pt, dinv, b2, pooled);
    fc_kernel<<<1, 32, 0, stream>>>(pooled, Wfc, bfc, out);
}
